// FNSD_51762945852046
// MI455X (gfx1250) — matrix operand order repaired and checked by reading, not compiled, not run on hardware
//
#include <hip/hip_runtime.h>
#include <hip/hip_bf16.h>
#include <stdint.h>

// ---------------------------------------------------------------------------
// GIN forward for MI455X (gfx1250, wave32, WMMA).
// GEMMs: bf16 A/B via v_wmma_f32_16x16x32_bf16 with fp32 accumulation.
// ---------------------------------------------------------------------------

typedef __attribute__((ext_vector_type(16))) __bf16 v16bf;
typedef __attribute__((ext_vector_type(8)))  float  v8f;

#define HD  128
#define LDW 136   // padded LDS pitch (bf16 elems): lane row-stride 272B -> no bank conflicts

// Native fptrunc: backend legalizes to hardware bf16 convert ops on gfx1250
// (packed v_cvt where possible), far cheaper than manual integer rounding.
static __device__ __forceinline__ __bf16 f2bf(float f) {
    return static_cast<__bf16>(f);
}

// C[n x 128] = A[n x 128] @ W[128 x 128] + bias, optional ReLU.
// Block: 256 threads = 8 waves; each wave computes a 16x128 strip (8 C tiles).
__global__ __launch_bounds__(256) void gemm128_bf16(
        const float* __restrict__ A, const float* __restrict__ W,
        const float* __restrict__ bias, float* __restrict__ C,
        int n, int relu)
{
    __shared__ __bf16 Wt[HD * LDW];           // transposed weights: Wt[col][k]
    const int t = threadIdx.x;

    // Stage + transpose W (fp32 -> bf16); 16384 elems / 256 threads = 64 each.
    for (int i = 0; i < 64; ++i) {
        int idx = i * 256 + t;
        int k = idx >> 7, c = idx & 127;
        Wt[c * LDW + k] = f2bf(W[idx]);
    }
    __syncthreads();

    const int wave = t >> 5;
    const int lane = t & 31;
    const int m    = lane & 15;
    const int half = lane >> 4;
    const int rowBase = blockIdx.x * 128 + wave * 16;
    int arow = rowBase + m; if (arow > n - 1) arow = n - 1;   // clamp tail reads
    const float* ap0 = A + (size_t)arow * HD;

    v8f acc[8] = {};
    #pragma unroll
    for (int ki = 0; ki < 4; ++ki) {
        const int k0 = ki * 32;
        // A fragment (16x32 bf16): lane half selects 16 consecutive K (matches B).
        union { v16bf v; __bf16 e[16]; } af;
        const float* ap = ap0 + k0 + half * 16;
        #pragma unroll
        for (int j = 0; j < 8; ++j) af.e[j]     = f2bf(ap[j]);        // K = k0+half*16+j
        #pragma unroll
        for (int j = 0; j < 8; ++j) af.e[8 + j] = f2bf(ap[8 + j]);    // K = k0+half*16+8+j
        #pragma unroll
        for (int nt = 0; nt < 8; ++nt) {
            // B fragment (32x16 bf16): lane = col, elems = 16 consecutive K.
            union { v16bf v; __bf16 e[16]; } bf;
            const __bf16* bp = &Wt[(nt * 16 + m) * LDW + k0 + half * 16];
            #pragma unroll
            for (int e2 = 0; e2 < 16; ++e2) bf.e[e2] = bp[e2];
            acc[nt] = __builtin_amdgcn_wmma_f32_16x16x32_bf16(
                false, af.v, false, bf.v, (short)0, acc[nt], false, false);
        }
    }
    // Epilogue: C/D layout — lanes 0-15: M=r, lanes 16-31: M=r+8, N=lane&15.
    #pragma unroll
    for (int nt = 0; nt < 8; ++nt) {
        const int col = nt * 16 + m;
        const float bv = bias[col];
        #pragma unroll
        for (int r = 0; r < 8; ++r) {
            int row = rowBase + half * 8 + r;
            if (row < n) {
                float v = acc[nt][r] + bv;
                if (relu) v = fmaxf(v, 0.0f);
                C[(size_t)row * HD + col] = v;
            }
        }
    }
}

__global__ void zero_f4(float4* __restrict__ p, long n4) {
    long stride = (long)gridDim.x * blockDim.x;
    for (long i = (long)blockIdx.x * blockDim.x + threadIdx.x; i < n4; i += stride)
        p[i] = make_float4(0.f, 0.f, 0.f, 0.f);
}

// Edge-parallel scatter-add: 32 lanes per edge, float4 granularity.
__global__ void scatter_edges(const float* __restrict__ h,
                              const int* __restrict__ erow, const int* __restrict__ ecol,
                              float* __restrict__ agg, int E)
{
    int tid = blockIdx.x * 256 + threadIdx.x;
    int e = tid >> 5;
    if (e >= 2 * E) return;
    int lane = tid & 31;
    int src, dst;
    if (e < E) { dst = erow[e];     src = ecol[e];     }   // (row <- col)
    else       { dst = ecol[e - E]; src = erow[e - E]; }   // flipped edge
    if (src == dst) return;                                // self-loop weight 0
    const float4 v = *(const float4*)(h + (size_t)src * HD + lane * 4);
    float* o = agg + (size_t)dst * HD + lane * 4;
    atomicAdd(o + 0, v.x); atomicAdd(o + 1, v.y);
    atomicAdd(o + 2, v.z); atomicAdd(o + 3, v.w);
}

// u = (1+eps)*h + agg   (in place over agg)
__global__ void gin_combine(const float* __restrict__ h, float* __restrict__ u,
                            const float* __restrict__ eps, long n4)
{
    const float alpha = 1.0f + eps[0];
    long stride = (long)gridDim.x * blockDim.x;
    for (long i = (long)blockIdx.x * blockDim.x + threadIdx.x; i < n4; i += stride) {
        float4 hv = ((const float4*)h)[i];
        float4 av = ((float4*)u)[i];
        av.x = alpha * hv.x + av.x; av.y = alpha * hv.y + av.y;
        av.z = alpha * hv.z + av.z; av.w = alpha * hv.w + av.w;
        ((float4*)u)[i] = av;
    }
}

// Deterministic two-stage column stats (sum, sumsq) over N rows.
__global__ void colstats_part(const float* __restrict__ z,
                              float* __restrict__ psum, float* __restrict__ psq, int n)
{
    __shared__ float ls[256], lq[256];
    int t = threadIdx.x;
    int col = t & 127, ro = t >> 7;
    int P = gridDim.x;
    int chunk = (n + P - 1) / P;
    int r0 = blockIdx.x * chunk;
    int r1 = r0 + chunk; if (r1 > n) r1 = n;
    float s = 0.f, q = 0.f;
    for (int r = r0 + ro; r < r1; r += 2) {
        float v = z[(size_t)r * HD + col];
        s += v; q += v * v;
    }
    ls[t] = s; lq[t] = q;
    __syncthreads();
    if (ro == 0) {
        psum[blockIdx.x * HD + col] = ls[t] + ls[t + 128];
        psq [blockIdx.x * HD + col] = lq[t] + lq[t + 128];
    }
}

// Fold partials in fixed order -> per-column affine (a, c): bn(x) = a*x + c.
__global__ void colstats_final(const float* __restrict__ psum, const float* __restrict__ psq,
                               const float* __restrict__ g, const float* __restrict__ b,
                               float* __restrict__ coef, int n, int P)
{
    int c = threadIdx.x;
    float s = 0.f, q = 0.f;
    for (int p = 0; p < P; ++p) { s += psum[p * HD + c]; q += psq[p * HD + c]; }
    float inv_n = 1.0f / (float)n;
    float mean = s * inv_n;
    float var  = q * inv_n - mean * mean;
    float rstd = rsqrtf(var + 1e-5f);
    float a = rstd * g[c];
    coef[c]      = a;
    coef[HD + c] = b[c] - mean * a;
}

__global__ void bn_relu(float* __restrict__ z, const float* __restrict__ coef, long n4)
{
    long stride = (long)gridDim.x * blockDim.x;
    for (long i = (long)blockIdx.x * blockDim.x + threadIdx.x; i < n4; i += stride) {
        int cb = (int)(i & 31) * 4;
        float4 v = ((float4*)z)[i];
        v.x = fmaxf(v.x * coef[cb + 0] + coef[HD + cb + 0], 0.f);
        v.y = fmaxf(v.y * coef[cb + 1] + coef[HD + cb + 1], 0.f);
        v.z = fmaxf(v.z * coef[cb + 2] + coef[HD + cb + 2], 0.f);
        v.w = fmaxf(v.w * coef[cb + 3] + coef[HD + cb + 3], 0.f);
        ((float4*)z)[i] = v;
    }
}

// h = LayerNorm(h + z) rowwise; one row per wave32, shfl_xor tree reduce.
__global__ __launch_bounds__(256) void residual_ln(
        float* __restrict__ h, const float* __restrict__ z,
        const float* __restrict__ g, const float* __restrict__ b, int n)
{
    int wave = threadIdx.x >> 5, lane = threadIdx.x & 31;
    int row = blockIdx.x * 8 + wave;
    if (row >= n) return;
    int c = lane * 4;
    float4 hv = *(const float4*)(h + (size_t)row * HD + c);
    float4 zv = *(const float4*)(z + (size_t)row * HD + c);
    hv.x += zv.x; hv.y += zv.y; hv.z += zv.z; hv.w += zv.w;
    float s = hv.x + hv.y + hv.z + hv.w;
    float q = hv.x * hv.x + hv.y * hv.y + hv.z * hv.z + hv.w * hv.w;
    #pragma unroll
    for (int off = 16; off > 0; off >>= 1) {
        s += __shfl_xor(s, off, 32);
        q += __shfl_xor(q, off, 32);
    }
    const float inv = 1.0f / 128.0f;
    float mean = s * inv;
    float var  = q * inv - mean * mean;
    float rstd = rsqrtf(var + 1e-5f);
    float4 o;
    o.x = (hv.x - mean) * rstd * g[c + 0] + b[c + 0];
    o.y = (hv.y - mean) * rstd * g[c + 1] + b[c + 1];
    o.z = (hv.z - mean) * rstd * g[c + 2] + b[c + 2];
    o.w = (hv.w - mean) * rstd * g[c + 3] + b[c + 3];
    *(float4*)(h + (size_t)row * HD + c) = o;
}

// Global mean pool: batch is sorted -> one block per graph, binary-search range.
__global__ void pool_mean(const float* __restrict__ h, const int* __restrict__ batch,
                          float* __restrict__ out, int n)
{
    int gId = blockIdx.x, c = threadIdx.x;
    int lo = 0, hi = n;
    while (lo < hi) { int mid = (lo + hi) >> 1; if (batch[mid] < gId) lo = mid + 1; else hi = mid; }
    int start = lo;
    hi = n;
    while (lo < hi) { int mid = (lo + hi) >> 1; if (batch[mid] < gId + 1) lo = mid + 1; else hi = mid; }
    int end = lo;
    float acc = 0.f;
    for (int r = start; r < end; ++r) acc += h[(size_t)r * HD + c];
    float cnt = (float)(end - start);
    out[gId * HD + c] = acc / fmaxf(cnt, 1.0f);
}

extern "C" void kernel_launch(void* const* d_in, const int* in_sizes, int n_in,
                              void* d_out, int out_size, void* d_ws, size_t ws_size,
                              hipStream_t stream)
{
    const float* x    = (const float*)d_in[0];
    const float* W0   = (const float*)d_in[1];
    const float* b0   = (const float*)d_in[2];
    const float* eps  = (const float*)d_in[3];
    const float* W1   = (const float*)d_in[4];
    const float* b1   = (const float*)d_in[5];
    const float* bn_g = (const float*)d_in[6];
    const float* bn_b = (const float*)d_in[7];
    const float* W2   = (const float*)d_in[8];
    const float* b2   = (const float*)d_in[9];
    const float* ln_g = (const float*)d_in[10];
    const float* ln_b = (const float*)d_in[11];
    const int*   ei   = (const int*)d_in[12];
    const int*   batch= (const int*)d_in[13];

    const int n = in_sizes[0] / HD;
    const int E = in_sizes[12] / 2;
    const int G = out_size / HD;
    const int L = 4;
    const int P = 512;

    float* ws   = (float*)d_ws;
    float* h    = ws;                              // [n, 128]
    float* bufA = ws + (size_t)n * HD;             // agg / u / z2
    float* bufB = ws + 2 * (size_t)n * HD;         // z
    float* psum = ws + 3 * (size_t)n * HD;         // [P, 128]
    float* psq  = psum + (size_t)P * HD;           // [P, 128]
    float* coef = psq  + (size_t)P * HD;           // [2, 128]

    const long n4 = (long)n * (HD / 4);
    dim3 blk(256);
    dim3 gemmGrid((n + 127) / 128);
    dim3 ewGrid(2048);
    dim3 scGrid((unsigned)(((long)2 * E * 32 + 255) / 256));

    // h = x @ W0 + b0
    gemm128_bf16<<<gemmGrid, blk, 0, stream>>>(x, W0, b0, h, n, 0);

    for (int i = 0; i < L; ++i) {
        // agg = scatter_add(h[col] -> row) over symmetrized edges
        zero_f4<<<ewGrid, blk, 0, stream>>>((float4*)bufA, n4);
        scatter_edges<<<scGrid, blk, 0, stream>>>(h, ei, ei + E, bufA, E);
        // u = (1+eps_i)*h + agg
        gin_combine<<<ewGrid, blk, 0, stream>>>(h, bufA, eps + i, n4);
        // z = u @ W1 + b1
        gemm128_bf16<<<gemmGrid, blk, 0, stream>>>(bufA, W1 + (size_t)i * HD * HD,
                                                   b1 + i * HD, bufB, n, 0);
        // BatchNorm (training stats) + ReLU
        colstats_part <<<dim3(P), blk, 0, stream>>>(bufB, psum, psq, n);
        colstats_final<<<dim3(1), dim3(HD), 0, stream>>>(psum, psq, bn_g + i * HD,
                                                         bn_b + i * HD, coef, n, P);
        bn_relu<<<ewGrid, blk, 0, stream>>>(bufB, coef, n4);
        // z2 = relu(z @ W2 + b2)  (ReLU fused in epilogue)
        gemm128_bf16<<<gemmGrid, blk, 0, stream>>>(bufB, W2 + (size_t)i * HD * HD,
                                                   b2 + i * HD, bufA, n, 1);
        // h = LayerNorm(h + z2)
        residual_ln<<<dim3((n + 7) / 8), blk, 0, stream>>>(h, bufA, ln_g + i * HD,
                                                           ln_b + i * HD, n);
    }
    pool_mean<<<dim3(G), dim3(HD), 0, stream>>>(h, batch, (float*)d_out, n);
}
